// SlidingWindowAttention_63075889709227
// MI455X (gfx1250) — matrix operand order repaired and checked by reading, not compiled, not run on hardware
//
#include <hip/hip_runtime.h>

// ---------------------------------------------------------------------------
// Sliding-window attention (W=256, non-causal) for MI455X / gfx1250, wave32.
// One workgroup (512 thr = 16 waves) handles one (batch*head, query-block).
// Each wave owns a 16-row query tile; workgroup streams 12 key tiles of 64
// keys through LDS (f16). Flash-style online softmax in base-2 (log2e folded
// into the Q scale); QK^T, P-rowsum and PV all on v_wmma_f32_16x16x32_f16
// with fp32 accumulation. Key axis is permuted in-tile (P and V identically)
// so P staging is packed ds_store_b64. Per-wave band relevance skip +
// edge-only masking keep VALU work off the interior tiles.
// ---------------------------------------------------------------------------

typedef __attribute__((ext_vector_type(16))) _Float16 v16h;
typedef __attribute__((ext_vector_type(8)))  _Float16 v8h;
typedef __attribute__((ext_vector_type(4)))  _Float16 v4h;
typedef __attribute__((ext_vector_type(8)))  float    v8f;

__device__ __forceinline__ v16h cat16(v8h lo, v8h hi) {
  return __builtin_shufflevector(lo, hi, 0,1,2,3,4,5,6,7,8,9,10,11,12,13,14,15);
}

__device__ __forceinline__ v8f wmma_f16(v16h a, v16h b, v8f c) {
  // D = A(16x32 f16) * B(32x16 f16) + C(16x16 f32)
  return __builtin_amdgcn_wmma_f32_16x16x32_f16(
      /*neg_a=*/false, a, /*neg_b=*/false, b,
      /*c_mod=*/(short)0, c, /*reuse_a=*/false, /*reuse_b=*/false);
}

__device__ __forceinline__ float redmax16(float x) {
  x = fmaxf(x, __shfl_xor(x, 1, 32));
  x = fmaxf(x, __shfl_xor(x, 2, 32));
  x = fmaxf(x, __shfl_xor(x, 4, 32));
  x = fmaxf(x, __shfl_xor(x, 8, 32));
  return x;
}

__global__ __launch_bounds__(512, 1)
void swa_wmma_kernel(const float* __restrict__ Qg,
                     const float* __restrict__ Kg,
                     const float* __restrict__ Vg,
                     float* __restrict__ Og)
{
  constexpr int S  = 4096;
  constexpr int D  = 64;
  constexpr int W  = 256;
  constexpr int NB = S / W;          // 16 query blocks
  constexpr int KT = 64;             // keys per tile
  constexpr float NEG = -1e30f;      // (base-2 domain; behavior identical)

  const int tid  = threadIdx.x;
  const int bh   = blockIdx.x >> 4;  // batch*head
  const int n    = blockIdx.x & (NB - 1);

  const int lane = tid & 31;
  const int wv   = tid >> 5;         // wave id == query M-tile (0..15)
  const int l15  = lane & 15;
  const int hlf  = lane >> 4;        // 0/1 lane half

  // LDS: K tile row-major [key][dim], V tile transposed + key-permuted
  // [dim][p(key)], per-wave P tile [row][p(key)]. 16B-aligned fragment runs.
  __shared__ __align__(16) _Float16 sK[64][80];
  __shared__ __align__(16) _Float16 sV[64][72];
  __shared__ __align__(16) _Float16 sP[16][16][64];

  const size_t baseQ = ((size_t)bh * S + (size_t)n * W) * D;

  // ---- Load Q fragments (A-layout); fold 1/sqrt(D) and log2(e) in ---------
  const float qs = 0.125f * 1.44269504088896340736f;   // base-2 softmax
  const int   qrow = wv * 16 + l15;
  v16h aQ[2];
#pragma unroll
  for (int kc = 0; kc < 2; ++kc) {
    const float* qp = Qg + baseQ + (size_t)qrow * D + kc * 32 + hlf * 8;
    float4 q0 = *(const float4*)(qp);
    float4 q1 = *(const float4*)(qp + 4);
    float4 q2 = *(const float4*)(qp + 16);
    float4 q3 = *(const float4*)(qp + 20);
    v16h a;
    a[0]  = (_Float16)(q0.x * qs); a[1]  = (_Float16)(q0.y * qs);
    a[2]  = (_Float16)(q0.z * qs); a[3]  = (_Float16)(q0.w * qs);
    a[4]  = (_Float16)(q1.x * qs); a[5]  = (_Float16)(q1.y * qs);
    a[6]  = (_Float16)(q1.z * qs); a[7]  = (_Float16)(q1.w * qs);
    a[8]  = (_Float16)(q2.x * qs); a[9]  = (_Float16)(q2.y * qs);
    a[10] = (_Float16)(q2.z * qs); a[11] = (_Float16)(q2.w * qs);
    a[12] = (_Float16)(q3.x * qs); a[13] = (_Float16)(q3.y * qs);
    a[14] = (_Float16)(q3.z * qs); a[15] = (_Float16)(q3.w * qs);
    aQ[kc] = a;
  }

  // Ones B-fragment: rowsum(P) via WMMA instead of lane shuffles.
  v16h bOnes;
#pragma unroll
  for (int i = 0; i < 16; ++i) bOnes[i] = (_Float16)1.0f;

  // ---- Accumulators / softmax state ---------------------------------------
  v8f acc[4];
  v8f z8 = {0.f,0.f,0.f,0.f,0.f,0.f,0.f,0.f};
#pragma unroll
  for (int nd = 0; nd < 4; ++nd) acc[nd] = z8;
  float m_run[8], l_run[8];
#pragma unroll
  for (int r = 0; r < 8; ++r) { m_run[r] = NEG; l_run[r] = 0.f; }

  // ---- Key-tile loop bounds (skip fully out-of-range boundary tiles) ------
  const int t0 = (n == 0)      ? 4 : 0;    // 12 tiles of 64 cover the 3W strip
  const int t1 = (n == NB - 1) ? 8 : 12;

  // staging: 512 threads x two float4 each for K and V (64 keys x 64 dims)
  const int skey = tid >> 3;          // 0..63
  const int sd   = (tid & 7) * 8;     // 0..56
  const int pkey = (skey & 15) * 4 + (skey >> 4);  // permuted key column
  const int kbase0 = n * W - W;       // first key of the 3W strip (global)

  float4 kr0, kr1, vr0, vr1;
  {
    int kg  = kbase0 + t0 * KT + skey;
    int kgc = kg < 0 ? 0 : (kg >= S ? S - 1 : kg);
    bool ok = (kg >= 0) && (kg < S);
    const float* kp = Kg + ((size_t)bh * S + kgc) * D + sd;
    const float* vp = Vg + ((size_t)bh * S + kgc) * D + sd;
    float4 zz = make_float4(0.f, 0.f, 0.f, 0.f);
    kr0 = ok ? *(const float4*)kp       : zz;
    kr1 = ok ? *(const float4*)(kp + 4) : zz;
    vr0 = ok ? *(const float4*)vp       : zz;
    vr1 = ok ? *(const float4*)(vp + 4) : zz;
  }

  const int qlo = wv * 16;            // block-relative first query row

  for (int t = t0; t < t1; ++t) {
    __syncthreads();                  // prior compute finished reading LDS
    // stage current tile: K row-major; V transposed with permuted key column
    {
      v8h hk;
      hk[0] = (_Float16)kr0.x; hk[1] = (_Float16)kr0.y;
      hk[2] = (_Float16)kr0.z; hk[3] = (_Float16)kr0.w;
      hk[4] = (_Float16)kr1.x; hk[5] = (_Float16)kr1.y;
      hk[6] = (_Float16)kr1.z; hk[7] = (_Float16)kr1.w;
      *(v8h*)&sK[skey][sd] = hk;
      sV[sd + 0][pkey] = (_Float16)vr0.x;
      sV[sd + 1][pkey] = (_Float16)vr0.y;
      sV[sd + 2][pkey] = (_Float16)vr0.z;
      sV[sd + 3][pkey] = (_Float16)vr0.w;
      sV[sd + 4][pkey] = (_Float16)vr1.x;
      sV[sd + 5][pkey] = (_Float16)vr1.y;
      sV[sd + 6][pkey] = (_Float16)vr1.z;
      sV[sd + 7][pkey] = (_Float16)vr1.w;
    }
    __syncthreads();

    // prefetch next tile's global data while this tile computes
    if (t + 1 < t1) {
      int kg  = kbase0 + (t + 1) * KT + skey;
      int kgc = kg < 0 ? 0 : (kg >= S ? S - 1 : kg);
      bool ok = (kg >= 0) && (kg < S);
      const float* kp = Kg + ((size_t)bh * S + kgc) * D + sd;
      const float* vp = Vg + ((size_t)bh * S + kgc) * D + sd;
      float4 zz = make_float4(0.f, 0.f, 0.f, 0.f);
      kr0 = ok ? *(const float4*)kp       : zz;
      kr1 = ok ? *(const float4*)(kp + 4) : zz;
      vr0 = ok ? *(const float4*)vp       : zz;
      vr1 = ok ? *(const float4*)(vp + 4) : zz;
    }

    // ---- per-wave band relevance (wave-uniform branch) --------------------
    const int klo = t * KT - W;       // block-relative first key of this tile
    const bool relevant = (klo + (KT - 1) >= qlo - W) && (klo <= qlo + 15 + W);
    if (relevant) {
      // ---- S = Q K^T : 4 key sub-tiles of 16 ------------------------------
      v8f c[4];
#pragma unroll
      for (int nt = 0; nt < 4; ++nt) {
        const _Float16* p0 = &sK[nt * 16 + l15][hlf * 8];
        const _Float16* p1 = &sK[nt * 16 + l15][32 + hlf * 8];
        v16h b0 = cat16(*(const v8h*)p0, *(const v8h*)(p0 + 16));
        v16h b1 = cat16(*(const v8h*)p1, *(const v8h*)(p1 + 16));
        c[nt] = wmma_f16(aQ[1], b1, wmma_f16(aQ[0], b0, z8));
      }

      // ---- band mask only for tiles straddling the band edge --------------
      const bool fullValid = (klo >= qlo + 15 - W) && (klo + (KT - 1) <= qlo + W);
      if (!fullValid) {
        const int qbase = qlo + 8 * hlf;
#pragma unroll
        for (int nt = 0; nt < 4; ++nt) {
          const int ki = klo + nt * 16 + l15;   // block-relative key col
#pragma unroll
          for (int r = 0; r < 8; ++r) {
            const int dd = ki - (qbase + r);
            const bool ok = (dd >= -W) && (dd <= W);
            c[nt][r] = ok ? c[nt][r] : NEG;
          }
        }
      }

      // ---- online softmax (base-2): max tree + exp2, packed P store -------
      float alpha[8];
#pragma unroll
      for (int r = 0; r < 8; ++r) {
        float rm = fmaxf(fmaxf(c[0][r], c[1][r]), fmaxf(c[2][r], c[3][r]));
        rm = redmax16(rm);
        float mn = fmaxf(m_run[r], rm);
        alpha[r] = __builtin_amdgcn_exp2f(m_run[r] - mn);  // 1.0 while NEG
        m_run[r] = mn;
        v4h pe;
#pragma unroll
        for (int nt = 0; nt < 4; ++nt)
          pe[nt] = (_Float16)__builtin_amdgcn_exp2f(c[nt][r] - mn);
        // permuted column p = l15*4 + nt  ->  one packed 8B store per row
        *(v4h*)&sP[wv][r + 8 * hlf][l15 * 4] = pe;
#pragma unroll
        for (int nd = 0; nd < 4; ++nd) acc[nd][r] *= alpha[r];
      }

      // ---- P: C-layout f32 -> A-layout f16 via per-wave LDS tile ----------
      v16h aP[2];
#pragma unroll
      for (int kc = 0; kc < 2; ++kc) {
        const _Float16* p = &sP[wv][l15][kc * 32 + hlf * 8];
        aP[kc] = cat16(*(const v8h*)p, *(const v8h*)(p + 16));
      }

      // ---- row sums of P with two WMMAs (no lane shuffles) ----------------
      v8f rsum = wmma_f16(aP[1], bOnes, wmma_f16(aP[0], bOnes, z8));
#pragma unroll
      for (int r = 0; r < 8; ++r)
        l_run[r] = l_run[r] * alpha[r] + rsum[r];

      // ---- O += P V : B-fragments from permuted, transposed sV ------------
#pragma unroll
      for (int nd = 0; nd < 4; ++nd) {
#pragma unroll
        for (int kc = 0; kc < 2; ++kc) {
          const _Float16* p = &sV[nd * 16 + l15][kc * 32 + hlf * 8];
          v16h bV = cat16(*(const v8h*)p, *(const v8h*)(p + 16));
          acc[nd] = wmma_f16(aP[kc], bV, acc[nd]);
        }
      }
    }
  }

  // ---- epilogue: normalize and store fp32 ---------------------------------
#pragma unroll
  for (int r = 0; r < 8; ++r) l_run[r] = 1.0f / l_run[r];
  float* op = Og + baseQ;
#pragma unroll
  for (int nd = 0; nd < 4; ++nd)
#pragma unroll
    for (int r = 0; r < 8; ++r)
      op[(size_t)(wv * 16 + r + 8 * hlf) * D + nd * 16 + l15] =
          acc[nd][r] * l_run[r];
}

extern "C" void kernel_launch(void* const* d_in, const int* in_sizes, int n_in,
                              void* d_out, int out_size, void* d_ws, size_t ws_size,
                              hipStream_t stream) {
  (void)n_in; (void)out_size; (void)d_ws; (void)ws_size;
  const float* Q = (const float*)d_in[0];
  const float* K = (const float*)d_in[1];
  const float* V = (const float*)d_in[2];
  float*       O = (float*)d_out;

  constexpr int S = 4096, D = 64, NB = 16;
  const int BH = in_sizes[0] / (S * D);      // batch*heads = 32
  dim3 grid(BH * NB), block(512);
  swa_wmma_kernel<<<grid, block, 0, stream>>>(Q, K, V, O);
}
